// MambaGraphEEG2TEXT_61529701483010
// MI455X (gfx1250) — compile-verified, matrix operand order checked by reading, not run on hardware
//
#include <hip/hip_runtime.h>

// ---------------------------------------------------------------------------
// Problem constants (match reference)
// ---------------------------------------------------------------------------
#define R_      10
#define C_      6
#define HID_    128
#define HEADS_  4
#define NL_     3
#define B_      16
#define T_      800
#define BT_     (B_ * T_)              // 12800
#define NTOT_   (BT_ * R_ * HID_)      // 16,384,000 floats per nf buffer
#define NEG_SLOPE 0.2f

#define APITCH  136                    // A-tile LDS pitch in halfs (16B row skew)
#define XPITCH  516                    // xp LDS pitch in floats (bank spread)
#define MAXE    64

typedef __bf16 bf16_t;
typedef bf16_t v16bf __attribute__((ext_vector_type(16)));
typedef float  v8f   __attribute__((ext_vector_type(8)));

union Frag {
    v16bf v;
    uint4 q[2];
    unsigned short u[16];
};

__device__ __forceinline__ unsigned short f2bf(float f) {
    // round-to-nearest-even f32 -> bf16
    unsigned u = __float_as_uint(f);
    unsigned r = u + 0x7FFFu + ((u >> 16) & 1u);
    return (unsigned short)(r >> 16);
}

__device__ __forceinline__ float gelu_exact(float x) {
    return 0.5f * x * (1.0f + erff(x * 0.70710678118654752f));
}

// ---------------------------------------------------------------------------
// Kernel 1: convert Wg (L,128,512) f32 -> bf16 pre-swizzled into WMMA
// B-fragment order: [l][nt(0..31)][s(0..3)][lane(0..31)][half 0..15].
// Lane L<16 holds column n=nt*16+L, K = s*32 + {0..7, 16..23};
// lane L>=16 holds same column, K = s*32 + {8..15, 24..31}.
// ---------------------------------------------------------------------------
__global__ void prep_wg_bf16(const float* __restrict__ Wg,
                             unsigned short* __restrict__ out) {
    int t = blockIdx.x * blockDim.x + threadIdx.x;
    if (t >= NL_ * 32 * 4 * 32) return;
    int lane = t & 31;
    int s    = (t >> 5) & 3;
    int nt   = (t >> 7) & 31;
    int l    = t >> 12;
    int kbase = s * 32;
    int n     = nt * 16 + (lane & 15);
    int khi   = (lane >> 4) ? 8 : 0;
    const float* W = Wg + (size_t)l * HID_ * (HEADS_ * HID_);
    unsigned short* o = out + (size_t)t * 16;
#pragma unroll
    for (int i = 0; i < 16; ++i) {
        int k = kbase + khi + ((i < 8) ? i : (8 + i));  // i>=8 -> +16+(i-8)
        o[i] = f2bf(W[k * (HEADS_ * HID_) + n]);
    }
}

// ---------------------------------------------------------------------------
// Kernel 2: per-region encoder  Linear(6->128) + LayerNorm + exact GELU.
// One wave32 per (b,t,r) node; 4 hidden units per lane; shuffle reduction.
// ---------------------------------------------------------------------------
__global__ __launch_bounds__(256)
void encoder_kernel(const float* __restrict__ x,     const float* __restrict__ Wenc,
                    const float* __restrict__ benc,  const float* __restrict__ gamma,
                    const float* __restrict__ beta,
                    float* __restrict__ region_out,  float* __restrict__ nf0) {
    int wave = threadIdx.x >> 5;
    int lane = threadIdx.x & 31;
    int g = blockIdx.x * 8 + wave;
    if (g >= BT_ * R_) return;
    int bt = g / R_, r = g - bt * R_;

    const float* xv = x + (size_t)bt * (R_ * C_) + r * C_;
    float xc[C_];
#pragma unroll
    for (int c = 0; c < C_; ++c) xc[c] = xv[c];

    float h[4], sum = 0.f, sumsq = 0.f;
#pragma unroll
    for (int i = 0; i < 4; ++i) {
        int hid = lane + 32 * i;
        float acc = benc[r * HID_ + hid];
#pragma unroll
        for (int c = 0; c < C_; ++c)
            acc = fmaf(xc[c], Wenc[(r * C_ + c) * HID_ + hid], acc);
        h[i] = acc; sum += acc; sumsq = fmaf(acc, acc, sumsq);
    }
#pragma unroll
    for (int m = 1; m < 32; m <<= 1) {
        sum   += __shfl_xor(sum,   m, 32);
        sumsq += __shfl_xor(sumsq, m, 32);
    }
    float mean = sum * (1.0f / HID_);
    float var  = sumsq * (1.0f / HID_) - mean * mean;
    float inv  = rsqrtf(var + 1e-5f);

    size_t base = (size_t)bt * (R_ * HID_) + r * HID_;
#pragma unroll
    for (int i = 0; i < 4; ++i) {
        int hid = lane + 32 * i;
        float hn = (h[i] - mean) * inv * gamma[r * HID_ + hid] + beta[r * HID_ + hid];
        float gv = gelu_exact(hn);
        region_out[base + hid] = gv;   // region_feats output
        nf0[base + hid]        = gv;   // nf for layer 0
    }
}

// ---------------------------------------------------------------------------
// Kernel 3: one fused GAT layer. One block per (b,t).
//   xp = nf(16x128 bf16, rows>=10 zero) @ W(128x512 bf16)  via v_wmma bf16
//   then attention logits, segment softmax by dst, head-mean aggregation,
//   bias + GELU + residual, all with xp resident in LDS.
// xp is padded to 16 rows so the WMMA result scatter is branch-free
// (rows 10..15 are zero garbage that is never read).
// ---------------------------------------------------------------------------
__global__ __launch_bounds__(256, 4)
void gat_layer_kernel(const float* __restrict__ nf_in, float* __restrict__ nf_out,
                      const unsigned short* __restrict__ WgBf,
                      const float* __restrict__ asrc, const float* __restrict__ adst,
                      const float* __restrict__ bg,
                      const int* __restrict__ src, const int* __restrict__ dst, int E) {
    __shared__ __align__(16) unsigned short ldsA[16 * APITCH];
    __shared__ float xp[16 * XPITCH];          // 16 rows: unconditional stores
    __shared__ float als[R_ * HEADS_], ald[R_ * HEADS_];
    __shared__ float alpha[MAXE * HEADS_];
    __shared__ int   es[MAXE], ed[MAXE];

    int tid = threadIdx.x;
    int bt  = blockIdx.x;
    const float* nfin = nf_in + (size_t)bt * (R_ * HID_);

    // stage A tile: f32 -> bf16, rows 10..15 zero-padded
    for (int idx = tid; idx < 16 * HID_; idx += 256) {
        int row = idx >> 7, k = idx & 127;
        float v = (row < R_) ? nfin[row * HID_ + k] : 0.0f;
        ldsA[row * APITCH + k] = f2bf(v);
    }
    if (tid < E) { es[tid] = src[tid]; ed[tid] = dst[tid]; }
    __syncthreads();

    // ---- WMMA GEMM: xp(16x512) = A(16x128) x W(128x512) ----
    int wave = tid >> 5, lane = tid & 31;
    int row16 = lane & 15;
    int khi   = (lane & 16) ? 8 : 0;

    Frag a[4];
#pragma unroll
    for (int s = 0; s < 4; ++s) {
        const unsigned short* ap = &ldsA[row16 * APITCH + s * 32 + khi];
        a[s].q[0] = *(const uint4*)(ap);        // K = s*32+khi   .. +7
        a[s].q[1] = *(const uint4*)(ap + 16);   // K = s*32+khi+16.. +23
    }
#pragma unroll
    for (int j = 0; j < 4; ++j) {
        int nt = wave * 4 + j;                  // N tile (16 cols)
        v8f acc = {};
#pragma unroll
        for (int s = 0; s < 4; ++s) {
            Frag b;
            const unsigned short* bp = WgBf + ((size_t)(nt * 4 + s) * 32 + lane) * 16;
            b.q[0] = *(const uint4*)(bp);
            b.q[1] = *(const uint4*)(bp + 8);
            acc = __builtin_amdgcn_wmma_f32_16x16x32_bf16(
                false, a[s].v, false, b.v, (short)0, acc, false, false);
        }
        int col = nt * 16 + row16;
        int rlo = (lane & 16) ? 8 : 0;          // C/D layout: vgpr j = row j (+8 hi lanes)
#pragma unroll
        for (int jj = 0; jj < 8; ++jj) {
            xp[(jj + rlo) * XPITCH + col] = acc[jj];   // branch-free scatter
        }
    }
    __syncthreads();

    // ---- attention logits: al_s/al_d[r][h] = <xp[r,h,:], a_src/dst[h,:]> ----
    if (tid < 2 * R_ * HEADS_) {
        int which = tid >= R_ * HEADS_;
        int t2 = tid - which * R_ * HEADS_;
        int r = t2 >> 2, hh = t2 & 3;
        const float* av = (which ? adst : asrc) + hh * HID_;
        const float* xr = &xp[r * XPITCH + hh * HID_];
        float acc = 0.f;
        for (int k = 0; k < HID_; ++k) acc = fmaf(xr[k], av[k], acc);
        if (which) ald[t2] = acc; else als[t2] = acc;
    }
    __syncthreads();

    // ---- segment softmax over edges grouped by destination ----
    if (tid < R_ * HEADS_) {
        int r = tid >> 2, hh = tid & 3;
        float m = -1e30f;
        for (int e = 0; e < E; ++e)
            if (ed[e] == r) {
                float v = als[es[e] * 4 + hh] + ald[r * 4 + hh];
                v = (v > 0.f) ? v : NEG_SLOPE * v;
                m = fmaxf(m, v);
            }
        float ssum = 0.f;
        for (int e = 0; e < E; ++e)
            if (ed[e] == r) {
                float v = als[es[e] * 4 + hh] + ald[r * 4 + hh];
                v = (v > 0.f) ? v : NEG_SLOPE * v;
                float ex = __expf(v - m);
                alpha[e * 4 + hh] = ex;
                ssum += ex;
            }
        float rs = 1.0f / ssum;                 // self-loops guarantee >=1 edge
        for (int e = 0; e < E; ++e)
            if (ed[e] == r) alpha[e * 4 + hh] *= rs;
    }
    __syncthreads();

    // ---- aggregate, mean over heads, bias, GELU, residual ----
    for (int o = tid; o < R_ * HID_; o += 256) {
        int r = o >> 7, f = o & 127;
        float acc = 0.f;
        for (int e = 0; e < E; ++e) {
            if (ed[e] == r) {
                const float* xs = &xp[es[e] * XPITCH + f];
#pragma unroll
                for (int hh = 0; hh < HEADS_; ++hh)
                    acc = fmaf(alpha[e * 4 + hh], xs[hh * HID_], acc);
            }
        }
        float outv = acc * (1.0f / HEADS_) + bg[f];
        nf_out[(size_t)bt * (R_ * HID_) + o] = gelu_exact(outv) + nfin[o];
    }
}

// ---------------------------------------------------------------------------
// Launch: prep weights once, encoder, then 3 GAT layers ping-ponging between
// workspace nf buffer and d_out's graph_features region (final layer lands
// in d_out). region_feats go to d_out's second half.
// ---------------------------------------------------------------------------
extern "C" void kernel_launch(void* const* d_in, const int* in_sizes, int n_in,
                              void* d_out, int out_size, void* d_ws, size_t ws_size,
                              hipStream_t stream) {
    (void)n_in; (void)out_size; (void)ws_size;
    const float* x     = (const float*)d_in[0];
    const float* Wenc  = (const float*)d_in[1];
    const float* benc  = (const float*)d_in[2];
    const float* gamma = (const float*)d_in[3];
    const float* beta  = (const float*)d_in[4];
    const float* Wg    = (const float*)d_in[5];
    const float* asrc  = (const float*)d_in[6];
    const float* adst  = (const float*)d_in[7];
    const float* bg    = (const float*)d_in[8];
    const int*   src   = (const int*)d_in[9];
    const int*   dst   = (const int*)d_in[10];
    int E = in_sizes[9];
    if (E > MAXE) E = MAXE;

    float* out_graph  = (float*)d_out;                 // (B,T,R*HID)
    float* out_region = out_graph + (size_t)NTOT_;     // (B,T,R,HID)
    float* nf_ws      = (float*)d_ws;                  // ping buffer
    unsigned short* WgBf = (unsigned short*)(nf_ws + NTOT_);

    prep_wg_bf16<<<(NL_ * 32 * 4 * 32 + 255) / 256, 256, 0, stream>>>(Wg, WgBf);
    encoder_kernel<<<(BT_ * R_) / 8, 256, 0, stream>>>(x, Wenc, benc, gamma, beta,
                                                       out_region, nf_ws);

    const size_t wl = (size_t)HID_ * HEADS_ * HID_;    // 65536 halfs / layer
    const int as = HEADS_ * HID_;                      // a_src/a_dst layer stride
    gat_layer_kernel<<<BT_, 256, 0, stream>>>(nf_ws, out_graph, WgBf + 0 * wl,
                                              asrc + 0 * as, adst + 0 * as,
                                              bg + 0 * HID_, src, dst, E);
    gat_layer_kernel<<<BT_, 256, 0, stream>>>(out_graph, nf_ws, WgBf + 1 * wl,
                                              asrc + 1 * as, adst + 1 * as,
                                              bg + 1 * HID_, src, dst, E);
    gat_layer_kernel<<<BT_, 256, 0, stream>>>(nf_ws, out_graph, WgBf + 2 * wl,
                                              asrc + 2 * as, adst + 2 * as,
                                              bg + 2 * HID_, src, dst, E);
}
